// DeformableTransformerEncoderLayer_37254546325717
// MI455X (gfx1250) — compile-verified, matrix operand order checked by reading, not run on hardware
//
#include <hip/hip_runtime.h>
#include <cstdint>
#include <cstddef>

// ---------------------------------------------------------------------------
// Deformable Transformer Encoder Layer for MI455X (gfx1250, wave32, WMMA).
// GEMMs: v_wmma_f32_16x16x32_f16, fp32 accumulate. Tiles staged to LDS as
// packed f16 (v_cvt_pk_rtz_f16_f32 -> ds_store_b128), software-pipelined so
// next slab's global loads overlap the current slab's WMMAs. N/K are template
// constants so B-tile loads use immediate offsets (no 64-bit add chains).
// Deformable core: one wave32 per (b,q,h), lane == channel (HD==32).
// ---------------------------------------------------------------------------

typedef __attribute__((ext_vector_type(16))) _Float16 v16h;
typedef __attribute__((ext_vector_type(8)))  float    v8f;

// Problem constants (match reference)
constexpr int Bc   = 4;
constexpr int Dc   = 256;
constexpr int Hc   = 8;
constexpr int LVc  = 4;
constexpr int Pc   = 4;
constexpr int DFFN = 1024;
constexpr int Lc   = 100*100 + 50*50 + 25*25 + 13*13; // 13294
constexpr int MR   = Bc * Lc;                         // 53176 rows

union FragH { v16h h; uint32_t u[8]; };

__device__ __forceinline__ int imin(int a, int b) { return a < b ? a : b; }
__device__ __forceinline__ int imax(int a, int b) { return a > b ? a : b; }

// Pack two f32 -> one dword of f16x2 (v_cvt_pk_rtz_f16_f32).
__device__ __forceinline__ uint32_t pk2(float a, float b) {
  typedef __fp16 h2v __attribute__((ext_vector_type(2)));
  union { h2v h; uint32_t u; } r;
  r.h = __builtin_amdgcn_cvt_pkrtz(a, b);
  return r.u;
}

// ---------------------------------------------------------------------------
// WMMA GEMM: C[M,N] = A[M,K] * Bw[K,N] + bias (optional ReLU)
// Block: 256 threads (8 waves). Macro-tile 64x64, K-step 32.
// Wave w: M-subtile (w&3)*16, N-strip (w>>2)*32 (two 16x16 accumulators).
// LDS row stride 20 dwords: conflict-free fragment reads AND 16B alignment
// so staging stores / fragment loads are b128. N,K compile-time -> B loads
// fold row strides into the 24-bit instruction offset.
// ---------------------------------------------------------------------------
template<int N, int K, bool RELU>
__global__ void __launch_bounds__(256) wmma_gemm_kernel(
    const float* __restrict__ A, const float* __restrict__ Bw,
    const float* __restrict__ bias, float* __restrict__ C, int M)
{
  constexpr int RSU = 20; // LDS row stride in dwords (40 halves)
  __shared__ uint32_t AsU[64 * RSU];
  __shared__ uint32_t BsU[64 * RSU];

  const int tid  = threadIdx.x;
  const int wv   = tid >> 5;
  const int lane = tid & 31;
  const int wm   = wv & 3;     // M sub-tile
  const int wn   = wv >> 2;    // N half (0/1)
  const int m0   = blockIdx.y * 64;
  const int n0   = blockIdx.x * 64;
  const int hg   = lane >> 4;  // K half-group per ISA 16-bit layout
  const int rc   = lane & 15;  // row (A) / col (B) within 16-wide tile

  // Staging map: A -> thread covers row=tid>>2, k-segment (tid&3)*8 (8 floats)
  //              B -> thread covers col=tid&63, k-segment (tid>>6)*8 (8 floats)
  const int arow = tid >> 2;
  const int aseg = tid & 3;
  const int grow = m0 + arow;
  const int crow = (grow < M) ? grow : (M - 1);     // clamped; zeroed below
  const size_t abase = (size_t)crow * K + aseg * 8;
  const int bn  = tid & 63;
  const int bks = tid >> 6;
  const size_t bbase = (size_t)(bks * 8) * N + (n0 + bn);

  float4 ra0, ra1;
  float  rb[8];

  // Prologue: load first K-slab into registers.
  ra0 = *(const float4*)(A + abase);
  ra1 = *(const float4*)(A + abase + 4);
  #pragma unroll
  for (int j = 0; j < 8; ++j) rb[j] = Bw[bbase + (size_t)j * N];
  if (grow >= M) { ra0 = make_float4(0,0,0,0); ra1 = make_float4(0,0,0,0); }

  v8f acc0 = {};
  v8f acc1 = {};

  for (int k0 = 0; k0 < K; k0 += 32) {
    // Publish staged registers to LDS as packed f16 (b128 stores).
    {
      uint32_t* ap = AsU + arow * RSU + aseg * 4;
      ap[0] = pk2(ra0.x, ra0.y);
      ap[1] = pk2(ra0.z, ra0.w);
      ap[2] = pk2(ra1.x, ra1.y);
      ap[3] = pk2(ra1.z, ra1.w);
      uint32_t* bp = BsU + bn * RSU + bks * 4;
      bp[0] = pk2(rb[0], rb[1]);
      bp[1] = pk2(rb[2], rb[3]);
      bp[2] = pk2(rb[4], rb[5]);
      bp[3] = pk2(rb[6], rb[7]);
    }
    __syncthreads();

    // Issue next slab's global loads now; they overlap the WMMAs below.
    if (k0 + 32 < K) {
      ra0 = *(const float4*)(A + abase + k0 + 32);
      ra1 = *(const float4*)(A + abase + k0 + 36);
      #pragma unroll
      for (int j = 0; j < 8; ++j)
        rb[j] = Bw[bbase + (size_t)(k0 + 32) * N + (size_t)j * N];
      if (grow >= M) { ra0 = make_float4(0,0,0,0); ra1 = make_float4(0,0,0,0); }
    }

    // Fragment gather per ISA 7.12.2: vgpr v holds K-pair
    // kp = (v<4 ? v : v+4) + 4*hg  -> two contiguous 4-dword groups (b128).
    FragH fa, fb0, fb1;
    {
      const uint32_t* fap  = AsU + (16 * wm + rc)      * RSU + 4 * hg;
      const uint32_t* fbp0 = BsU + (32 * wn + rc)      * RSU + 4 * hg;
      const uint32_t* fbp1 = BsU + (32 * wn + 16 + rc) * RSU + 4 * hg;
      #pragma unroll
      for (int j = 0; j < 4; ++j) {
        fa.u[j]      = fap[j];
        fa.u[4 + j]  = fap[8 + j];
        fb0.u[j]     = fbp0[j];
        fb0.u[4 + j] = fbp0[8 + j];
        fb1.u[j]     = fbp1[j];
        fb1.u[4 + j] = fbp1[8 + j];
      }
    }

    acc0 = __builtin_amdgcn_wmma_f32_16x16x32_f16(
        false, fa.h, false, fb0.h, (short)0, acc0, false, false);
    acc1 = __builtin_amdgcn_wmma_f32_16x16x32_f16(
        false, fa.h, false, fb1.h, (short)0, acc1, false, false);
    __syncthreads();
  }

  // Epilogue: C/D layout -> vgpr rr: row = rr + 8*hg, col = lane&15.
  #pragma unroll
  for (int rr = 0; rr < 8; ++rr) {
    int orow = m0 + 16 * wm + rr + 8 * hg;
    if (orow < M) {
      int gc0 = n0 + 32 * wn + rc;
      float o0 = acc0[rr] + bias[gc0];
      float o1 = acc1[rr] + bias[gc0 + 16];
      if (RELU) { o0 = fmaxf(o0, 0.0f); o1 = fmaxf(o1, 0.0f); }
      C[(size_t)orow * N + gc0]      = o0;
      C[(size_t)orow * N + gc0 + 16] = o1;
    }
  }
}

// ---------------------------------------------------------------------------
// q = src + pos (float4 vectorized; n is a multiple of 4)
// ---------------------------------------------------------------------------
__global__ void __launch_bounds__(256) add_kernel(
    const float4* __restrict__ a, const float4* __restrict__ b,
    float4* __restrict__ o, int n4)
{
  int i = blockIdx.x * 256 + threadIdx.x;
  if (i < n4) {
    float4 x = a[i], y = b[i];
    o[i] = make_float4(x.x + y.x, x.y + y.y, x.z + y.z, x.w + y.w);
  }
}

// ---------------------------------------------------------------------------
// softmax over LV*P=16 per (b,q,h). attn_raw is [MR,128]; thread t covers the
// contiguous 16 logits at t*16 (== (row*H+h)*16).
// ---------------------------------------------------------------------------
__global__ void __launch_bounds__(256) softmax_aw_kernel(
    const float* __restrict__ attn_raw, float* __restrict__ aw, int nrh)
{
  int t = blockIdx.x * 256 + threadIdx.x;
  if (t >= nrh) return;
  const float4* p4 = (const float4*)(attn_raw + (size_t)t * 16);
  float v[16];
  #pragma unroll
  for (int i = 0; i < 4; ++i) {
    float4 q = p4[i];
    v[4*i+0] = q.x; v[4*i+1] = q.y; v[4*i+2] = q.z; v[4*i+3] = q.w;
  }
  float mx = v[0];
  #pragma unroll
  for (int j = 1; j < 16; ++j) mx = fmaxf(mx, v[j]);
  float s = 0.0f;
  #pragma unroll
  for (int j = 0; j < 16; ++j) { v[j] = __expf(v[j] - mx); s += v[j]; }
  float inv = 1.0f / s;
  float4* o4 = (float4*)(aw + (size_t)t * 16);
  #pragma unroll
  for (int i = 0; i < 4; ++i)
    o4[i] = make_float4(v[4*i+0]*inv, v[4*i+1]*inv, v[4*i+2]*inv, v[4*i+3]*inv);
}

// ---------------------------------------------------------------------------
// loc = reference_points + off / (W_l, H_l). One thread per (b,q,h,lv,p).
// ---------------------------------------------------------------------------
__global__ void __launch_bounds__(256) loc_kernel(
    const float* __restrict__ off_raw, const float* __restrict__ refp,
    const int* __restrict__ shapes, float* __restrict__ loc, int n)
{
  int t = blockIdx.x * 256 + threadIdx.x;
  if (t >= n) return;                 // n = MR * H*LV*P (=128)
  int j  = t & 127;                   // (h*LV+lv)*P+p
  int rq = t >> 7;                    // b*L + q
  int lv = (j >> 2) & 3;
  float2 off = *(const float2*)(off_raw + (size_t)rq * 256 + j * 2);
  int Hl = shapes[2 * lv + 0];
  int Wl = shapes[2 * lv + 1];
  float2 rp = *(const float2*)(refp + ((size_t)rq * 4 + lv) * 2);
  float2 out;
  out.x = rp.x + off.x / (float)Wl;
  out.y = rp.y + off.y / (float)Hl;
  *(float2*)(loc + (size_t)t * 2) = out;
}

// ---------------------------------------------------------------------------
// Deformable attention core. One wave32 per (b,q,h); lane == channel (HD=32).
// Value gathers are 32 consecutive floats -> one coalesced 128B line per tap.
// ---------------------------------------------------------------------------
__global__ void __launch_bounds__(256) deform_kernel(
    const float* __restrict__ value, const float* __restrict__ loc,
    const float* __restrict__ aw, const int* __restrict__ shapes,
    const int* __restrict__ lsi, float* __restrict__ out)
{
  int gw   = (blockIdx.x * 256 + threadIdx.x) >> 5;
  int lane = threadIdx.x & 31;
  if (gw >= MR * Hc) return;
  int h  = gw & 7;
  int rq = gw >> 3;                   // b*L + q
  int bL = (rq / Lc) * Lc;            // b*L

  float acc = 0.0f;
  #pragma unroll
  for (int lv = 0; lv < LVc; ++lv) {
    int Hl = shapes[2 * lv + 0];
    int Wl = shapes[2 * lv + 1];
    int start = lsi[lv];
    #pragma unroll
    for (int p = 0; p < Pc; ++p) {
      size_t pt = (((size_t)rq * Hc + h) * LVc + lv) * Pc + p;
      float2 lxy = *(const float2*)(loc + pt * 2);
      float a  = aw[pt];
      float px = lxy.x * (float)Wl - 0.5f;
      float py = lxy.y * (float)Hl - 0.5f;
      float x0f = floorf(px), y0f = floorf(py);
      float fx = px - x0f, fy = py - y0f;
      int x0 = (int)x0f, y0 = (int)y0f;
      #pragma unroll
      for (int dy = 0; dy < 2; ++dy) {
        #pragma unroll
        for (int dx = 0; dx < 2; ++dx) {
          int xi = x0 + dx, yi = y0 + dy;
          float w = (dx ? fx : 1.0f - fx) * (dy ? fy : 1.0f - fy);
          bool valid = (xi >= 0) && (xi < Wl) && (yi >= 0) && (yi < Hl);
          int xc = imin(imax(xi, 0), Wl - 1);
          int yc = imin(imax(yi, 0), Hl - 1);
          int l  = start + yc * Wl + xc;
          float g = value[(size_t)(bL + l) * 256 + h * 32 + lane];
          acc += (valid ? w * a : 0.0f) * g;
        }
      }
    }
  }
  out[(size_t)rq * 256 + h * 32 + lane] = acc;
}

// ---------------------------------------------------------------------------
// out = LayerNorm(a + r) * g + be  -- one 256-thread block per row (D=256).
// ---------------------------------------------------------------------------
__global__ void __launch_bounds__(256) ln_kernel(
    const float* __restrict__ a, const float* __restrict__ r,
    const float* __restrict__ g, const float* __restrict__ be,
    float* __restrict__ o)
{
  __shared__ float red[256];
  int row = blockIdx.x, c = threadIdx.x;
  size_t base = (size_t)row * 256;
  float v = a[base + c] + r[base + c];
  red[c] = v;
  __syncthreads();
  #pragma unroll
  for (int s = 128; s > 0; s >>= 1) {
    if (c < s) red[c] += red[c + s];
    __syncthreads();
  }
  float mean = red[0] * (1.0f / 256.0f);
  __syncthreads();
  float d = v - mean;
  red[c] = d * d;
  __syncthreads();
  #pragma unroll
  for (int s = 128; s > 0; s >>= 1) {
    if (c < s) red[c] += red[c + s];
    __syncthreads();
  }
  float var = red[0] * (1.0f / 256.0f);
  o[base + c] = d * rsqrtf(var + 1e-5f) * g[c] + be[c];
}

// ---------------------------------------------------------------------------
extern "C" void kernel_launch(void* const* d_in, const int* in_sizes, int n_in,
                              void* d_out, int out_size, void* d_ws, size_t ws_size,
                              hipStream_t stream)
{
  const float* src  = (const float*)d_in[0];
  const float* pos  = (const float*)d_in[1];
  const float* refp = (const float*)d_in[2];
  const int*   shp  = (const int*)d_in[3];
  const int*   lsi  = (const int*)d_in[4];
  const float* Wv   = (const float*)d_in[5];
  const float* bv   = (const float*)d_in[6];
  const float* Woff = (const float*)d_in[7];
  const float* boff = (const float*)d_in[8];
  const float* Wa   = (const float*)d_in[9];
  const float* ba   = (const float*)d_in[10];
  const float* Wout = (const float*)d_in[11];
  const float* bout = (const float*)d_in[12];
  const float* W1   = (const float*)d_in[13];
  const float* b1   = (const float*)d_in[14];
  const float* W2   = (const float*)d_in[15];
  const float* b2   = (const float*)d_in[16];
  const float* g1   = (const float*)d_in[17];
  const float* be1  = (const float*)d_in[18];
  const float* g2   = (const float*)d_in[19];
  const float* be2  = (const float*)d_in[20];

  // Output layout: x [MR*256] | loc [MR*256] | aw [MR*128]
  float* x_out   = (float*)d_out;
  float* loc_out = x_out + (size_t)MR * 256;
  float* aw_out  = loc_out + (size_t)MR * 256;

  // Workspace carve-up (buffers reused once their producer is consumed):
  //   q_buf   -> later src2      (dead after off/attn GEMMs)
  //   val_buf -> later ffn_raw   (dead after deform)
  //   off_buf -> later attn_out  (dead after loc kernel)
  float* ws      = (float*)d_ws;
  float* q_buf   = ws;                               // MR*256
  float* val_buf = q_buf   + (size_t)MR * 256;       // MR*256
  float* off_buf = val_buf + (size_t)MR * 256;       // MR*256
  float* atr_buf = off_buf + (size_t)MR * 256;       // MR*128
  float* x1_buf  = atr_buf + (size_t)MR * 128;       // MR*256
  float* hid_buf = x1_buf  + (size_t)MR * 256;       // MR*1024

  const int MBLK = (MR + 63) / 64;                   // 831
  dim3 blk(256);

  // 1) q = src + pos
  {
    int n4 = MR * Dc / 4;
    add_kernel<<<(n4 + 255) / 256, blk, 0, stream>>>(
        (const float4*)src, (const float4*)pos, (float4*)q_buf, n4);
  }
  // 2) value = src @ W_value + b_value
  wmma_gemm_kernel<256, 256, false><<<dim3(4, MBLK), blk, 0, stream>>>(
      src, Wv, bv, val_buf, MR);
  // 3) off_raw = q @ W_off + b_off   (N = 256)
  wmma_gemm_kernel<256, 256, false><<<dim3(4, MBLK), blk, 0, stream>>>(
      q_buf, Woff, boff, off_buf, MR);
  // 4) attn_raw = q @ W_attn + b_attn (N = 128)
  wmma_gemm_kernel<128, 256, false><<<dim3(2, MBLK), blk, 0, stream>>>(
      q_buf, Wa, ba, atr_buf, MR);
  // 5) aw = softmax(attn_raw) over 16
  {
    int nrh = MR * Hc;
    softmax_aw_kernel<<<(nrh + 255) / 256, blk, 0, stream>>>(atr_buf, aw_out, nrh);
  }
  // 6) loc = ref + off / (W,H)
  {
    int n = MR * Hc * LVc * Pc;
    loc_kernel<<<(n + 255) / 256, blk, 0, stream>>>(off_buf, refp, shp, loc_out, n);
  }
  // 7) deformable gather core -> attn_out (reuses off_buf)
  deform_kernel<<<MR, blk, 0, stream>>>(val_buf, loc_out, aw_out, shp, lsi, off_buf);
  // 8) src2 = attn_out @ W_out + b_out (reuses q_buf)
  wmma_gemm_kernel<256, 256, false><<<dim3(4, MBLK), blk, 0, stream>>>(
      off_buf, Wout, bout, q_buf, MR);
  // 9) x1 = LN(src + src2)
  ln_kernel<<<MR, blk, 0, stream>>>(src, q_buf, g1, be1, x1_buf);
  // 10) hidden = relu(x1 @ W1 + b1)
  wmma_gemm_kernel<DFFN, 256, true><<<dim3(16, MBLK), blk, 0, stream>>>(
      x1_buf, W1, b1, hid_buf, MR);
  // 11) ffn_raw = hidden @ W2 + b2 (reuses val_buf)
  wmma_gemm_kernel<256, DFFN, false><<<dim3(4, MBLK), blk, 0, stream>>>(
      hid_buf, W2, b2, val_buf, MR);
  // 12) x = LN(x1 + ffn_raw)
  ln_kernel<<<MR, blk, 0, stream>>>(x1_buf, val_buf, g2, be2, x_out);
}
